// EGNNConv_79207786873522
// MI455X (gfx1250) — matrix operand order repaired and checked by reading, not compiled, not run on hardware
//
#include <hip/hip_runtime.h>
#include <hip/hip_bf16.h>
#include <math.h>

typedef _Float16 v4h  __attribute__((ext_vector_type(4)));
typedef _Float16 v8h  __attribute__((ext_vector_type(8)));
typedef _Float16 v16h __attribute__((ext_vector_type(16)));
typedef float    v8f  __attribute__((ext_vector_type(8)));

#define LDS_FENCE() asm volatile("" ::: "memory")

static __device__ __forceinline__ v16h cat8(v8h lo, v8h hi) {
    return __builtin_shufflevector(lo, hi, 0, 1, 2, 3, 4, 5, 6, 7, 8, 9, 10, 11, 12, 13, 14, 15);
}

static __device__ __forceinline__ v8f wmma16(v16h a, v16h b, v8f c) {
    // D(16x16 f32) = A(16x32 f16) * B(32x16 f16) + C
    return __builtin_amdgcn_wmma_f32_16x16x32_f16(false, a, false, b, (short)0, c, false, false);
}

static __device__ __forceinline__ float fsilu(float x) {
    return x / (1.0f + __expf(-x));
}

// ---------------------------------------------------------------------------
// Weight swizzle: row-major f32 W[Ktot][128]  ->  f16 B-operand tiles.
// Tile (kk,t) covers K=kk*32..+31, N=t*16..+15. Per tile: 32 lanes x 16 f16,
// lane l holds B[k][n] with n = l&15, k = (i>>3)*16 + (l>>4)*8 + (i&7)  (i = elem 0..15),
// stored lane-contiguous so the GEMM loads it with two b128 per lane.
// ---------------------------------------------------------------------------
__global__ void egnn_prep_kernel(const float* __restrict__ W, _Float16* __restrict__ dst, int Ktot) {
    int gid = blockIdx.x * blockDim.x + threadIdx.x;
    int total = Ktot * 128;
    if (gid >= total) return;
    int tileIdx = gid >> 9;     // 512 elems per tile
    int r       = gid & 511;
    int l       = r >> 4;       // lane
    int i       = r & 15;       // element within lane
    int kk      = tileIdx >> 3; // K-chunk
    int t       = tileIdx & 7;  // N-tile
    int k       = kk * 32 + (i >> 3) * 16 + (l >> 4) * 8 + (i & 7);
    int n       = t * 16 + (l & 15);
    dst[gid] = (_Float16)W[k * 128 + n];
}

// ---------------------------------------------------------------------------
// Node LayerNorm, f32 -> f16 output. One wave per node (128 feats, 4/lane).
// ---------------------------------------------------------------------------
__global__ __launch_bounds__(256) void egnn_ln_kernel(const float* __restrict__ h,
                                                      const float* __restrict__ g,
                                                      const float* __restrict__ b,
                                                      _Float16* __restrict__ hn16, int Nn) {
    int wv   = blockIdx.x * 8 + (threadIdx.x >> 5);
    int lane = threadIdx.x & 31;
    if (wv >= Nn) return;
    const float4* hp = (const float4*)(h + (size_t)wv * 128);
    float4 x = hp[lane];
    float  s = x.x + x.y + x.z + x.w;
#pragma unroll
    for (int m = 1; m < 32; m <<= 1) s += __shfl_xor(s, m, 32);
    float mu = s * (1.0f / 128.0f);
    float e0 = x.x - mu, e1 = x.y - mu, e2 = x.z - mu, e3 = x.w - mu;
    float ss = e0 * e0 + e1 * e1 + e2 * e2 + e3 * e3;
#pragma unroll
    for (int m = 1; m < 32; m <<= 1) ss += __shfl_xor(ss, m, 32);
    float inv = rsqrtf(ss * (1.0f / 128.0f) + 1e-5f);
    float4 g4 = ((const float4*)g)[lane];
    float4 b4 = ((const float4*)b)[lane];
    v4h o;
    o[0] = (_Float16)(e0 * inv * g4.x + b4.x);
    o[1] = (_Float16)(e1 * inv * g4.y + b4.y);
    o[2] = (_Float16)(e2 * inv * g4.z + b4.z);
    o[3] = (_Float16)(e3 * inv * g4.w + b4.w);
    *(v4h*)(hn16 + (size_t)wv * 128 + lane * 4) = o;
}

// ---------------------------------------------------------------------------
// Edge kernel: one wave processes a 16-edge tile through the whole edge MLP,
// LayerNorm, coord MLP, and all scatter-adds. 128 WMMAs per wave.
// C/D layout: VGPR j holds M=j (lanes 0-15) / M=j+8 (lanes 16-31), N=lane&15.
// ---------------------------------------------------------------------------
__global__ __launch_bounds__(256) void egnn_edge_kernel(
    const _Float16* __restrict__ hn16, const int* __restrict__ eidx, const float* __restrict__ coord,
    const _Float16* __restrict__ w1s, const float* __restrict__ w1last, const float* __restrict__ b1,
    const _Float16* __restrict__ w2s, const float* __restrict__ b2,
    const float* __restrict__ ln_g, const float* __restrict__ ln_b,
    const _Float16* __restrict__ cw1s, const float* __restrict__ cb1, const float* __restrict__ cw2,
    float* __restrict__ agg, float* __restrict__ tsum, int E) {
    __shared__ __attribute__((aligned(16))) _Float16 stage_all[8][2048]; // 16x128 f16 per wave
    __shared__ float edata_all[8][16][4];                                // dx,dy,dz,radial
    __shared__ float scale_all[8][16];

    const int w    = threadIdx.x >> 5;
    const int lane = threadIdx.x & 31;
    const int tile = blockIdx.x * 8 + w;
    const int e0   = tile * 16;
    if (e0 >= E) return;

    _Float16* stage   = stage_all[w];
    float (*edata)[4] = edata_all[w];
    float* lscale     = scale_all[w];

    const int nl = lane & 15;
    const int hh = lane >> 4;

    int rowv = 0, colv = 0;
    if (lane < 16) {
        int e = e0 + lane;
        if (e < E) { rowv = eidx[e]; colv = eidx[E + e]; }
        float ax = coord[rowv * 3 + 0], ay = coord[rowv * 3 + 1], az = coord[rowv * 3 + 2];
        float bx = coord[colv * 3 + 0], by = coord[colv * 3 + 1], bz = coord[colv * 3 + 2];
        float dx = ax - bx, dy = ay - by, dz = az - bz;
        edata[lane][0] = dx; edata[lane][1] = dy; edata[lane][2] = dz;
        edata[lane][3] = dx * dx + dy * dy + dz * dz;
    }
    LDS_FENCE(); // LDS is in-order within a wave; block compiler reordering only

    const int rA = __shfl(rowv, nl, 32); // A row M = lane&15 (both halves)
    const int cA = __shfl(colv, nl, 32);
    int rm8[8];
#pragma unroll
    for (int j = 0; j < 8; ++j) rm8[j] = __shfl(rowv, j + 8 * hh, 32); // scatter rows

    // ---- GEMM1: [hn[row] | hn[col]] (16x256) @ edge_w1[0:256] ----
    v8f acc[8] = {};
#pragma unroll
    for (int kk = 0; kk < 8; ++kk) {
        int nodeA = (kk < 4) ? rA : cA;
        const _Float16* ap = hn16 + (size_t)nodeA * 128 + (kk & 3) * 32 + hh * 8;
        v16h a = cat8(*(const v8h*)ap, *(const v8h*)(ap + 16));
        const _Float16* bp = w1s + (size_t)(kk * 8) * 512 + lane * 16;
#pragma unroll
        for (int t = 0; t < 8; ++t) {
            v16h bmat = cat8(*(const v8h*)bp, *(const v8h*)(bp + 8));
            acc[t] = wmma16(a, bmat, acc[t]);
            bp += 512;
        }
    }
    // radial rank-1 update (row 256 of edge_w1) + bias + SiLU
    float r8v[8];
#pragma unroll
    for (int j = 0; j < 8; ++j) r8v[j] = edata[j + 8 * hh][3];
#pragma unroll
    for (int t = 0; t < 8; ++t) {
        float wl = w1last[t * 16 + nl];
        float bb = b1[t * 16 + nl];
#pragma unroll
        for (int j = 0; j < 8; ++j) acc[t][j] = fsilu(acc[t][j] + r8v[j] * wl + bb);
    }
    // restage e1 as f16 row-major (C-layout -> A-layout transpose via LDS)
#pragma unroll
    for (int t = 0; t < 8; ++t)
#pragma unroll
        for (int j = 0; j < 8; ++j) stage[(j + 8 * hh) * 128 + t * 16 + nl] = (_Float16)acc[t][j];
    LDS_FENCE();

    // ---- GEMM2: e1 (16x128) @ edge_w2 ----
    v8f acc2[8] = {};
#pragma unroll
    for (int kk = 0; kk < 4; ++kk) {
        const _Float16* ap = stage + nl * 128 + kk * 32 + hh * 8;
        v16h a = cat8(*(const v8h*)ap, *(const v8h*)(ap + 16));
        const _Float16* bp = w2s + (size_t)(kk * 8) * 512 + lane * 16;
#pragma unroll
        for (int t = 0; t < 8; ++t) {
            v16h bmat = cat8(*(const v8h*)bp, *(const v8h*)(bp + 8));
            acc2[t] = wmma16(a, bmat, acc2[t]);
            bp += 512;
        }
    }
#pragma unroll
    for (int t = 0; t < 8; ++t) {
        float bb = b2[t * 16 + nl];
#pragma unroll
        for (int j = 0; j < 8; ++j) acc2[t][j] = fsilu(acc2[t][j] + bb);
    }

    // ---- edge LayerNorm: each row lives in one 16-lane half ----
    float mean[8], inv[8];
#pragma unroll
    for (int j = 0; j < 8; ++j) {
        float s = 0.0f, ss = 0.0f;
#pragma unroll
        for (int t = 0; t < 8; ++t) { float x = acc2[t][j]; s += x; ss += x * x; }
#pragma unroll
        for (int m = 1; m < 16; m <<= 1) { s += __shfl_xor(s, m, 32); ss += __shfl_xor(ss, m, 32); }
        float mu  = s * (1.0f / 128.0f);
        float var = ss * (1.0f / 128.0f) - mu * mu;
        mean[j] = mu;
        inv[j]  = rsqrtf(var + 1e-5f);
    }
#pragma unroll
    for (int t = 0; t < 8; ++t) {
        float g = ln_g[t * 16 + nl], bb = ln_b[t * 16 + nl];
#pragma unroll
        for (int j = 0; j < 8; ++j) acc2[t][j] = (acc2[t][j] - mean[j]) * inv[j] * g + bb;
    }

    // restage e3 for GEMM3 + scatter-add e3 into agg (f32 atomics, L2-resident)
#pragma unroll
    for (int t = 0; t < 8; ++t)
#pragma unroll
        for (int j = 0; j < 8; ++j) stage[(j + 8 * hh) * 128 + t * 16 + nl] = (_Float16)acc2[t][j];
    LDS_FENCE();
#pragma unroll
    for (int j = 0; j < 8; ++j) {
        if (e0 + j + 8 * hh < E) {
            float* dst = agg + (size_t)rm8[j] * 128 + nl;
#pragma unroll
            for (int t = 0; t < 8; ++t) atomicAdd(dst + t * 16, acc2[t][j]);
        }
    }

    // ---- GEMM3: e3 @ coord_w1, SiLU, then dot with coord_w2 -> scale ----
    v8f acc3[8] = {};
#pragma unroll
    for (int kk = 0; kk < 4; ++kk) {
        const _Float16* ap = stage + nl * 128 + kk * 32 + hh * 8;
        v16h a = cat8(*(const v8h*)ap, *(const v8h*)(ap + 16));
        const _Float16* bp = cw1s + (size_t)(kk * 8) * 512 + lane * 16;
#pragma unroll
        for (int t = 0; t < 8; ++t) {
            v16h bmat = cat8(*(const v8h*)bp, *(const v8h*)(bp + 8));
            acc3[t] = wmma16(a, bmat, acc3[t]);
            bp += 512;
        }
    }
    float cwv[8];
#pragma unroll
    for (int t = 0; t < 8; ++t) cwv[t] = cw2[t * 16 + nl];
#pragma unroll
    for (int j = 0; j < 8; ++j) {
        float p = 0.0f;
#pragma unroll
        for (int t = 0; t < 8; ++t) p += fsilu(acc3[t][j] + cb1[t * 16 + nl]) * cwv[t];
#pragma unroll
        for (int m = 1; m < 16; m <<= 1) p += __shfl_xor(p, m, 32);
        if (nl == 0) lscale[j + 8 * hh] = p;
    }
    LDS_FENCE();
    if (lane < 16 && (e0 + lane) < E) {
        float s  = lscale[lane];
        float* t = tsum + (size_t)rowv * 4;
        atomicAdd(t + 0, edata[lane][0] * s);
        atomicAdd(t + 1, edata[lane][1] * s);
        atomicAdd(t + 2, edata[lane][2] * s);
        atomicAdd(t + 3, 1.0f);
    }
}

// ---------------------------------------------------------------------------
// Node kernel: [hn | agg] (16x256) @ node_w1, SiLU, @ node_w2, +b2, +residual.
// ---------------------------------------------------------------------------
__global__ __launch_bounds__(256) void egnn_node_kernel(
    const _Float16* __restrict__ hn16, const float* __restrict__ agg,
    const _Float16* __restrict__ nw1s, const float* __restrict__ nb1,
    const _Float16* __restrict__ nw2s, const float* __restrict__ nb2,
    const float* __restrict__ h0, float* __restrict__ hout, int Nn) {
    __shared__ __attribute__((aligned(16))) _Float16 stage_all[8][2048];
    const int w    = threadIdx.x >> 5;
    const int lane = threadIdx.x & 31;
    const int nt   = blockIdx.x * 8 + w;
    const int n0   = nt * 16;
    if (n0 >= Nn) return;
    _Float16* stage = stage_all[w];
    const int nl = lane & 15;
    const int hh = lane >> 4;
    int rowA = n0 + nl;
    if (rowA >= Nn) rowA = Nn - 1; // clamp (stores are guarded)

    v8f acc[8] = {};
#pragma unroll
    for (int kk = 0; kk < 8; ++kk) {
        v16h a;
        if (kk < 4) {
            const _Float16* ap = hn16 + (size_t)rowA * 128 + (kk & 3) * 32 + hh * 8;
            a = cat8(*(const v8h*)ap, *(const v8h*)(ap + 16));
        } else {
            const float* fp = agg + (size_t)rowA * 128 + (kk & 3) * 32 + hh * 8;
            float4 f0 = *(const float4*)(fp);
            float4 f1 = *(const float4*)(fp + 4);
            float4 f2 = *(const float4*)(fp + 16);
            float4 f3 = *(const float4*)(fp + 20);
            a[0]  = (_Float16)f0.x; a[1]  = (_Float16)f0.y; a[2]  = (_Float16)f0.z; a[3]  = (_Float16)f0.w;
            a[4]  = (_Float16)f1.x; a[5]  = (_Float16)f1.y; a[6]  = (_Float16)f1.z; a[7]  = (_Float16)f1.w;
            a[8]  = (_Float16)f2.x; a[9]  = (_Float16)f2.y; a[10] = (_Float16)f2.z; a[11] = (_Float16)f2.w;
            a[12] = (_Float16)f3.x; a[13] = (_Float16)f3.y; a[14] = (_Float16)f3.z; a[15] = (_Float16)f3.w;
        }
        const _Float16* bp = nw1s + (size_t)(kk * 8) * 512 + lane * 16;
#pragma unroll
        for (int t = 0; t < 8; ++t) {
            v16h bmat = cat8(*(const v8h*)bp, *(const v8h*)(bp + 8));
            acc[t] = wmma16(a, bmat, acc[t]);
            bp += 512;
        }
    }
#pragma unroll
    for (int t = 0; t < 8; ++t) {
        float bb = nb1[t * 16 + nl];
#pragma unroll
        for (int j = 0; j < 8; ++j) stage[(j + 8 * hh) * 128 + t * 16 + nl] = (_Float16)fsilu(acc[t][j] + bb);
    }
    LDS_FENCE();

    v8f acc2[8] = {};
#pragma unroll
    for (int kk = 0; kk < 4; ++kk) {
        const _Float16* ap = stage + nl * 128 + kk * 32 + hh * 8;
        v16h a = cat8(*(const v8h*)ap, *(const v8h*)(ap + 16));
        const _Float16* bp = nw2s + (size_t)(kk * 8) * 512 + lane * 16;
#pragma unroll
        for (int t = 0; t < 8; ++t) {
            v16h bmat = cat8(*(const v8h*)bp, *(const v8h*)(bp + 8));
            acc2[t] = wmma16(a, bmat, acc2[t]);
            bp += 512;
        }
    }
#pragma unroll
    for (int t = 0; t < 8; ++t) {
        float bb = nb2[t * 16 + nl];
#pragma unroll
        for (int j = 0; j < 8; ++j) {
            int node = n0 + j + 8 * hh;
            if (node < Nn) {
                size_t idx = (size_t)node * 128 + t * 16 + nl;
                hout[idx]  = acc2[t][j] + bb + h0[idx];
            }
        }
    }
}

__global__ void egnn_coord_kernel(const float* __restrict__ coord, const float* __restrict__ tsum,
                                  float* __restrict__ cout, int Nn) {
    int n = blockIdx.x * blockDim.x + threadIdx.x;
    if (n >= Nn) return;
    float inv = 1.0f / fmaxf(tsum[n * 4 + 3], 1.0f);
    cout[n * 3 + 0] = coord[n * 3 + 0] + tsum[n * 4 + 0] * inv;
    cout[n * 3 + 1] = coord[n * 3 + 1] + tsum[n * 4 + 1] * inv;
    cout[n * 3 + 2] = coord[n * 3 + 2] + tsum[n * 4 + 2] * inv;
}

// ---------------------------------------------------------------------------
extern "C" void kernel_launch(void* const* d_in, const int* in_sizes, int n_in,
                              void* d_out, int out_size, void* d_ws, size_t ws_size,
                              hipStream_t stream) {
    const float* h          = (const float*)d_in[0];
    const int*   edge_index = (const int*)d_in[1];
    const float* coord      = (const float*)d_in[2];
    const float* node_ln_g  = (const float*)d_in[3];
    const float* node_ln_b  = (const float*)d_in[4];
    const float* edge_ln_g  = (const float*)d_in[5];
    const float* edge_ln_b  = (const float*)d_in[6];
    const float* edge_w1    = (const float*)d_in[7];
    const float* edge_b1    = (const float*)d_in[8];
    const float* edge_w2    = (const float*)d_in[9];
    const float* edge_b2    = (const float*)d_in[10];
    const float* node_w1    = (const float*)d_in[11];
    const float* node_b1    = (const float*)d_in[12];
    const float* node_w2    = (const float*)d_in[13];
    const float* node_b2    = (const float*)d_in[14];
    const float* coord_w1   = (const float*)d_in[15];
    const float* coord_b1   = (const float*)d_in[16];
    const float* coord_w2   = (const float*)d_in[17];

    const int N = in_sizes[0] / 128;
    const int E = in_sizes[1] / 2;

    // workspace carve-out (256B aligned regions)
    char*  ws  = (char*)d_ws;
    size_t off = 0;
    auto take  = [&](size_t bytes) -> char* {
        char* p = ws + off;
        off += (bytes + 255) & ~(size_t)255;
        return p;
    };
    _Float16* hn16 = (_Float16*)take((size_t)N * 128 * 2);
    float*    agg  = (float*)take((size_t)N * 128 * 4);
    float*    tsum = (float*)take((size_t)N * 16);
    _Float16* w1s  = (_Float16*)take(256 * 128 * 2);
    _Float16* w2s  = (_Float16*)take(128 * 128 * 2);
    _Float16* cw1s = (_Float16*)take(128 * 128 * 2);
    _Float16* nw1s = (_Float16*)take(256 * 128 * 2);
    _Float16* nw2s = (_Float16*)take(128 * 128 * 2);
    (void)ws_size; (void)n_in; (void)out_size;

    float* hout = (float*)d_out;
    float* cout = hout + (size_t)N * 128;

    // zero accumulators every call (graph-capturable)
    hipMemsetAsync(agg, 0, (size_t)N * 128 * 4, stream);
    hipMemsetAsync(tsum, 0, (size_t)N * 16, stream);

    // weight swizzle (tiny)
    egnn_prep_kernel<<<(256 * 128 + 255) / 256, 256, 0, stream>>>(edge_w1, w1s, 256);
    egnn_prep_kernel<<<(128 * 128 + 255) / 256, 256, 0, stream>>>(edge_w2, w2s, 128);
    egnn_prep_kernel<<<(128 * 128 + 255) / 256, 256, 0, stream>>>(coord_w1, cw1s, 128);
    egnn_prep_kernel<<<(256 * 128 + 255) / 256, 256, 0, stream>>>(node_w1, nw1s, 256);
    egnn_prep_kernel<<<(128 * 128 + 255) / 256, 256, 0, stream>>>(node_w2, nw2s, 128);

    // node LayerNorm -> f16
    egnn_ln_kernel<<<(N + 7) / 8, 256, 0, stream>>>(h, node_ln_g, node_ln_b, hn16, N);

    // fused edge pipeline
    int etiles = (E + 15) / 16;
    egnn_edge_kernel<<<(etiles + 7) / 8, 256, 0, stream>>>(
        hn16, edge_index, coord,
        w1s, edge_w1 + 256 * 128, edge_b1,
        w2s, edge_b2, edge_ln_g, edge_ln_b,
        cw1s, coord_b1, coord_w2,
        agg, tsum, E);

    // node MLP + residual
    int ntiles = (N + 15) / 16;
    egnn_node_kernel<<<(ntiles + 7) / 8, 256, 0, stream>>>(
        hn16, agg, nw1s, node_b1, nw2s, node_b2, h, hout, N);

    // coord update
    egnn_coord_kernel<<<(N + 255) / 256, 256, 0, stream>>>(coord, tsum, cout, N);
}
